// CrossAttention_42142219108973
// MI455X (gfx1250) — matrix operand order repaired and checked by reading, not compiled, not run on hardware
//
#include <hip/hip_runtime.h>
#include <hip/hip_bf16.h>

#define B_    8
#define L_    8192
#define S_    512
#define DQ_   256
#define DC_   768
#define H_    8
#define DH_   64
#define INNER_ 512
#define AWAVES_ 8   // waves per attention block

#if defined(__has_builtin)
#  if __has_builtin(__builtin_amdgcn_global_load_async_to_lds_b128) && \
      __has_builtin(__builtin_amdgcn_s_wait_asynccnt)
#    define USE_ASYNC_LDS 1
#  endif
#endif
#ifndef USE_ASYNC_LDS
#  define USE_ASYNC_LDS 0
#endif

typedef __attribute__((ext_vector_type(16))) _Float16 v16h;
typedef __attribute__((ext_vector_type(8)))  float    v8f;
typedef __attribute__((ext_vector_type(2)))  _Float16 h2;
typedef __attribute__((ext_vector_type(2)))  float    f2;

// ---------------------------------------------------------------------------
// WMMA wrapper: D(16x16 f32) = A(16x32 f16) * B(32x16 f16) + C
// ---------------------------------------------------------------------------
static __device__ inline v8f wmma_f16(v16h a, v16h b, v8f c) {
  return __builtin_amdgcn_wmma_f32_16x16x32_f16(
      /*neg_a=*/false, a, /*neg_b=*/false, b,
      /*c_mod=*/(short)0, c, /*reuse_a=*/false, /*reuse_b=*/false);
}

// ---------------------------------------------------------------------------
// Fragment loaders (per CDNA5 ISA 7.12.2 VGPR layouts, wave32)
// ---------------------------------------------------------------------------
static __device__ inline v16h load_a_f16(const _Float16* base, int stride, int lane) {
  const int m  = lane & 15;
  const int kb = (lane < 16) ? 0 : 8;
  const _Float16* rp = base + (size_t)m * stride;
  v16h a;
#pragma unroll
  for (int j = 0; j < 8; ++j) {
    const int k = kb + ((j < 4) ? (2 * j) : (16 + 2 * (j - 4)));
    h2 pr = *reinterpret_cast<const h2*>(rp + k);
    a[2 * j] = pr[0]; a[2 * j + 1] = pr[1];
  }
  return a;
}

static __device__ inline v16h load_a_f32(const float* base, int stride, int lane) {
  const int m  = lane & 15;
  const int kb = (lane < 16) ? 0 : 8;
  const float* rp = base + (size_t)m * stride;
  v16h a;
#pragma unroll
  for (int j = 0; j < 8; ++j) {
    const int k = kb + ((j < 4) ? (2 * j) : (16 + 2 * (j - 4)));
    f2 pr = *reinterpret_cast<const f2*>(rp + k);
    a[2 * j] = (_Float16)pr[0]; a[2 * j + 1] = (_Float16)pr[1];
  }
  return a;
}

// B fragment from K-contiguous storage: element (k,n) at base[n*stride + k]
static __device__ inline v16h load_b_kc(const _Float16* base, int stride, int lane) {
  const int n  = lane & 15;
  const int kb = (lane < 16) ? 0 : 8;
  const _Float16* cp = base + (size_t)n * stride + kb;
  v16h b;
#pragma unroll
  for (int j = 0; j < 8; ++j) {
    h2 pr = *reinterpret_cast<const h2*>(cp + ((j < 4) ? (2 * j) : (16 + 2 * (j - 4))));
    b[2 * j] = pr[0]; b[2 * j + 1] = pr[1];
  }
  return b;
}

#if USE_ASYNC_LDS
// Builtin parameter types (from hipcc diagnostic): pointers to 16B int vector,
// src in addrspace(1) (printed as __device__), dst in addrspace(3).
typedef int v4i __attribute__((ext_vector_type(4)));
typedef __attribute__((address_space(1))) v4i v4i_g;
typedef __attribute__((address_space(3))) v4i v4i_s;

// 16-byte async global -> LDS copy (per lane), tracked by ASYNCcnt
static __device__ inline void async_cp16(const _Float16* g, _Float16* l) {
  __builtin_amdgcn_global_load_async_to_lds_b128((v4i_g*)g, (v4i_s*)l, 0, 0);
}
#endif

// ---------------------------------------------------------------------------
// Weight transpose + fp32 -> f16: W[K][N] -> WT[N][K]
// ---------------------------------------------------------------------------
__global__ void wtrans_kernel(const float* __restrict__ W, _Float16* __restrict__ WT,
                              int K, int N) {
  int idx = blockIdx.x * blockDim.x + threadIdx.x;
  if (idx >= K * N) return;
  int k = idx / N, n = idx % N;
  WT[(size_t)n * K + k] = (_Float16)W[(size_t)k * N + n];
}

// ---------------------------------------------------------------------------
// Q projection: Qh[b*H+h][l][dh] = f16( x @ Wq ); 16x64 tile per wave
// ---------------------------------------------------------------------------
__global__ void __launch_bounds__(32)
qproj_kernel(const float* __restrict__ x, const _Float16* __restrict__ WqT,
             _Float16* __restrict__ Qh) {
  const int lane = threadIdx.x & 31;
  const int mt = blockIdx.x;                  // B*L/16 tiles
  const int nt = blockIdx.y;                  // INNER/64 tile groups
  const float* Ab = x + (size_t)mt * 16 * DQ_;
  v8f acc[4] = {{}, {}, {}, {}};
  for (int k0 = 0; k0 < DQ_; k0 += 32) {
    v16h a = load_a_f32(Ab + k0, DQ_, lane);
#pragma unroll
    for (int nn = 0; nn < 4; ++nn) {
      const _Float16* Bb = WqT + (size_t)(nt * 64 + nn * 16) * DQ_ + k0;
      v16h b = load_b_kc(Bb, DQ_, lane);
      acc[nn] = wmma_f16(a, b, acc[nn]);
    }
  }
  const int n    = lane & 15;
  const int moff = (lane < 16) ? 0 : 8;
#pragma unroll
  for (int nn = 0; nn < 4; ++nn) {
    const int ng = nt * 64 + nn * 16 + n;
    const int h = ng >> 6, dh = ng & 63;
#pragma unroll
    for (int j = 0; j < 8; ++j) {
      int row = mt * 16 + moff + j;           // in [0, B*L)
      int b = row / L_, l = row % L_;
      Qh[(((size_t)(b * H_ + h)) * L_ + l) * DH_ + dh] = (_Float16)acc[nn][j];
    }
  }
}

// ---------------------------------------------------------------------------
// K/V projection: Kh[bh][s][dh], Vt[bh][dh][s]; 16x64 tile per wave
// ---------------------------------------------------------------------------
__global__ void __launch_bounds__(32)
kvproj_kernel(const float* __restrict__ ctx,
              const _Float16* __restrict__ WkT, const _Float16* __restrict__ WvT,
              _Float16* __restrict__ Kh, _Float16* __restrict__ Vt) {
  const int lane = threadIdx.x & 31;
  const int mt = blockIdx.x;                  // B*S/16 tiles
  const int nt = blockIdx.y;                  // INNER/64 tile groups
  const int which = blockIdx.z;
  const float* Ab = ctx + (size_t)mt * 16 * DC_;
  const _Float16* WT = which ? WvT : WkT;
  v8f acc[4] = {{}, {}, {}, {}};
  for (int k0 = 0; k0 < DC_; k0 += 32) {
    v16h a = load_a_f32(Ab + k0, DC_, lane);
#pragma unroll
    for (int nn = 0; nn < 4; ++nn) {
      const _Float16* Bb = WT + (size_t)(nt * 64 + nn * 16) * DC_ + k0;
      v16h b = load_b_kc(Bb, DC_, lane);
      acc[nn] = wmma_f16(a, b, acc[nn]);
    }
  }
  const int n    = lane & 15;
  const int moff = (lane < 16) ? 0 : 8;
#pragma unroll
  for (int nn = 0; nn < 4; ++nn) {
    const int ng = nt * 64 + nn * 16 + n;
    const int h = ng >> 6, dh = ng & 63;
#pragma unroll
    for (int j = 0; j < 8; ++j) {
      int row = mt * 16 + moff + j;           // in [0, B*S)
      int b = row / S_, s = row % S_;
      size_t bh = (size_t)(b * H_ + h);
      if (which == 0)
        Kh[(bh * S_ + s) * DH_ + dh] = (_Float16)acc[nn][j];
      else
        Vt[(bh * DH_ + dh) * S_ + s] = (_Float16)acc[nn][j];
    }
  }
}

// ---------------------------------------------------------------------------
// Fused flash attention. Block = 8 waves = 8 query strips of 16 for one (b,h).
// K/V tiles staged into double-buffered LDS via async global->LDS copies
// (ASYNCcnt) when available; scores 4 WMMAs, online softmax, P through
// per-wave LDS, P*V 4 WMMAs.
// ---------------------------------------------------------------------------
__global__ void __launch_bounds__(32 * AWAVES_)
attn_kernel(const _Float16* __restrict__ Qh, const _Float16* __restrict__ Kh,
            const _Float16* __restrict__ Vt, _Float16* __restrict__ AO) {
  __shared__ __align__(16) _Float16 Ks[2][32 * 64];          // [buf][key][dh]
  __shared__ __align__(16) _Float16 Vs[2][64 * 32];          // [buf][dh][key]
  __shared__ __align__(16) _Float16 Plds[AWAVES_][16 * 32];  // per-wave P
  const int tid  = threadIdx.x;
  const int wave = tid >> 5;
  const int lane = tid & 31;
  const int bh = blockIdx.y;                  // b*H + h
  const int qt = blockIdx.x * AWAVES_ + wave; // query tile (L/16)
  const int b = bh / H_, h = bh % H_;
  const float scale = 0.125f;                 // 1/sqrt(DH)

  const _Float16* Qb = Qh + ((size_t)bh * L_ + (size_t)qt * 16) * DH_;
  const v16h aq0 = load_a_f16(Qb,      DH_, lane);   // dh 0..31
  const v16h aq1 = load_a_f16(Qb + 32, DH_, lane);   // dh 32..63

  v8f o0 = {}, o1 = {}, o2 = {}, o3 = {};
  float mrow[8], lrow[8];
#pragma unroll
  for (int j = 0; j < 8; ++j) { mrow[j] = -1e30f; lrow[j] = 0.0f; }

  const int n    = lane & 15;
  const int moff = (lane < 16) ? 0 : 8;
  const int dhc  = tid >> 2, qd = (tid & 3) * 8;    // V staging chunk

  const _Float16* Kg0 = Kh + (size_t)bh * S_ * DH_;  // contiguous [key][dh]
  const _Float16* Vg0 = Vt + (size_t)bh * DH_ * S_;  // [dh][S]
  const int NT = S_ / 32;

#if USE_ASYNC_LDS
  // prime buffer 0: K tile = 4KB contiguous (256 x 16B); V tile = 64 rows x 64B
  async_cp16(Kg0 + tid * 8, Ks[0] + tid * 8);
  async_cp16(Vg0 + (size_t)dhc * S_ + qd, Vs[0] + dhc * 32 + qd);
#endif

  for (int kt = 0; kt < NT; ++kt) {
#if USE_ASYNC_LDS
    const int cur = kt & 1;
    __builtin_amdgcn_s_wait_asynccnt(0);       // my async copies landed
    __syncthreads();                           // everyone's copies landed
    if (kt + 1 < NT) {                         // overlap next tile with compute
      async_cp16(Kg0 + (size_t)(kt + 1) * 32 * DH_ + tid * 8,
                 Ks[cur ^ 1] + tid * 8);
      async_cp16(Vg0 + (size_t)dhc * S_ + (kt + 1) * 32 + qd,
                 Vs[cur ^ 1] + dhc * 32 + qd);
    }
    const _Float16* KsC = Ks[cur];
    const _Float16* VsC = Vs[cur];
#else
    {
      const uint4* srcK = reinterpret_cast<const uint4*>(Kg0 + (size_t)kt * 32 * DH_);
      reinterpret_cast<uint4*>(Ks[0])[tid] = srcK[tid];
      const uint4* srcV = reinterpret_cast<const uint4*>(Vg0 + (size_t)dhc * S_ + kt * 32 + qd);
      *reinterpret_cast<uint4*>(Vs[0] + dhc * 32 + qd) = *srcV;
      if (kt + 1 < NT) {
        __builtin_prefetch(srcK + 256 + tid, 0, 1);  // next K tile
        __builtin_prefetch(srcV + 4, 0, 1);          // next V chunk (+32 keys)
      }
    }
    __syncthreads();
    const _Float16* KsC = Ks[0];
    const _Float16* VsC = Vs[0];
#endif

    // ---- scores: 2 key sub-tiles x 2 dh chunks ----
    v8f s0 = {}, s1 = {};
    { v16h bk = load_b_kc(KsC,                 DH_, lane); s0 = wmma_f16(aq0, bk, s0); }
    { v16h bk = load_b_kc(KsC + 32,            DH_, lane); s0 = wmma_f16(aq1, bk, s0); }
    { v16h bk = load_b_kc(KsC + 16 * DH_,      DH_, lane); s1 = wmma_f16(aq0, bk, s1); }
    { v16h bk = load_b_kc(KsC + 16 * DH_ + 32, DH_, lane); s1 = wmma_f16(aq1, bk, s1); }

    // ---- online softmax ----
    float alpha[8];
#pragma unroll
    for (int j = 0; j < 8; ++j) {
      s0[j] *= scale; s1[j] *= scale;
      float t = fmaxf(s0[j], s1[j]);
      t = fmaxf(t, __shfl_xor(t, 1));
      t = fmaxf(t, __shfl_xor(t, 2));
      t = fmaxf(t, __shfl_xor(t, 4));
      t = fmaxf(t, __shfl_xor(t, 8));         // stays within each 16-lane half
      float mnew = fmaxf(mrow[j], t);
      alpha[j] = __expf(mrow[j] - mnew);
      float p0 = __expf(s0[j] - mnew);
      float p1 = __expf(s1[j] - mnew);
      s0[j] = p0; s1[j] = p1;
      float r = p0 + p1;
      r += __shfl_xor(r, 1);
      r += __shfl_xor(r, 2);
      r += __shfl_xor(r, 4);
      r += __shfl_xor(r, 8);
      lrow[j] = lrow[j] * alpha[j] + r;
      mrow[j] = mnew;
      o0[j] *= alpha[j]; o1[j] *= alpha[j]; o2[j] *= alpha[j]; o3[j] *= alpha[j];
    }

    // ---- P: D-layout -> per-wave LDS -> A-layout (in-order within wave) ----
    _Float16* Pw = Plds[wave];
#pragma unroll
    for (int j = 0; j < 8; ++j) {
      Pw[(moff + j) * 32 + n]      = (_Float16)s0[j];
      Pw[(moff + j) * 32 + n + 16] = (_Float16)s1[j];
    }
    v16h ap = load_a_f16(Pw, 32, lane);

    // ---- P * V ----
    { v16h bv = load_b_kc(VsC + 0 * 16 * 32, 32, lane); o0 = wmma_f16(ap, bv, o0); }
    { v16h bv = load_b_kc(VsC + 1 * 16 * 32, 32, lane); o1 = wmma_f16(ap, bv, o1); }
    { v16h bv = load_b_kc(VsC + 2 * 16 * 32, 32, lane); o2 = wmma_f16(ap, bv, o2); }
    { v16h bv = load_b_kc(VsC + 3 * 16 * 32, 32, lane); o3 = wmma_f16(ap, bv, o3); }
#if !USE_ASYNC_LDS
    __syncthreads();                          // before next K/V overwrite
#endif
  }

#pragma unroll
  for (int j = 0; j < 8; ++j) {
    float inv = 1.0f / lrow[j];
    int l = qt * 16 + moff + j;
    size_t base = ((size_t)(b * L_ + l)) * INNER_ + (size_t)h * DH_;
    AO[base + 0 * 16 + n] = (_Float16)(o0[j] * inv);
    AO[base + 1 * 16 + n] = (_Float16)(o1[j] * inv);
    AO[base + 2 * 16 + n] = (_Float16)(o2[j] * inv);
    AO[base + 3 * 16 + n] = (_Float16)(o3[j] * inv);
  }
}

// ---------------------------------------------------------------------------
// Output projection + bias + seq_len mask: 16x64 tile per wave, fp32 out
// ---------------------------------------------------------------------------
__global__ void __launch_bounds__(32)
oproj_kernel(const _Float16* __restrict__ AO, const _Float16* __restrict__ WoT,
             const float* __restrict__ bo, const int* __restrict__ seq_lens,
             float* __restrict__ out) {
  const int lane = threadIdx.x & 31;
  const int mt = blockIdx.x;                  // B*L/16 tiles
  const int nt = blockIdx.y;                  // DQ/64 tile groups
  const _Float16* Ab = AO + (size_t)mt * 16 * INNER_;
  v8f acc[4] = {{}, {}, {}, {}};
  for (int k0 = 0; k0 < INNER_; k0 += 32) {
    v16h a = load_a_f16(Ab + k0, INNER_, lane);
#pragma unroll
    for (int nn = 0; nn < 4; ++nn) {
      const _Float16* Bb = WoT + (size_t)(nt * 64 + nn * 16) * INNER_ + k0;
      v16h b = load_b_kc(Bb, INNER_, lane);
      acc[nn] = wmma_f16(a, b, acc[nn]);
    }
  }
  const int n    = lane & 15;
  const int moff = (lane < 16) ? 0 : 8;
#pragma unroll
  for (int nn = 0; nn < 4; ++nn) {
    const int ng = nt * 64 + nn * 16 + n;
    const float bias = bo[ng];
#pragma unroll
    for (int j = 0; j < 8; ++j) {
      int row = mt * 16 + moff + j;
      int b = row / L_, l = row % L_;
      float v = acc[nn][j] + bias;
      out[(size_t)row * DQ_ + ng] = (l < seq_lens[b]) ? v : 0.0f;
    }
  }
}

// ---------------------------------------------------------------------------
// Host launcher
// ---------------------------------------------------------------------------
extern "C" void kernel_launch(void* const* d_in, const int* in_sizes, int n_in,
                              void* d_out, int out_size, void* d_ws, size_t ws_size,
                              hipStream_t stream) {
  (void)in_sizes; (void)n_in; (void)out_size; (void)ws_size;
  const float* x        = (const float*)d_in[0];
  const float* ctx      = (const float*)d_in[1];
  const int*   seq_lens = (const int*)  d_in[2];
  const float* Wq       = (const float*)d_in[3];
  const float* Wk       = (const float*)d_in[4];
  const float* Wv       = (const float*)d_in[5];
  const float* Wo       = (const float*)d_in[6];
  const float* bo       = (const float*)d_in[7];
  float* out            = (float*)d_out;

  char* ws = (char*)d_ws;
  size_t off = 0;
  auto alloc = [&](size_t bytes) -> void* {
    void* p = ws + off;
    off += (bytes + 255) & ~(size_t)255;
    return p;
  };
  _Float16* WqT = (_Float16*)alloc((size_t)INNER_ * DQ_ * 2);
  _Float16* WkT = (_Float16*)alloc((size_t)INNER_ * DC_ * 2);
  _Float16* WvT = (_Float16*)alloc((size_t)INNER_ * DC_ * 2);
  _Float16* WoT = (_Float16*)alloc((size_t)DQ_ * INNER_ * 2);
  _Float16* Qh  = (_Float16*)alloc((size_t)B_ * H_ * L_ * DH_ * 2);
  _Float16* Kh  = (_Float16*)alloc((size_t)B_ * H_ * S_ * DH_ * 2);
  _Float16* Vt  = (_Float16*)alloc((size_t)B_ * H_ * DH_ * S_ * 2);
  _Float16* AO  = (_Float16*)alloc((size_t)B_ * L_ * INNER_ * 2);

  // 1) weight conversion/transposition
  wtrans_kernel<<<(DQ_ * INNER_ + 255) / 256, 256, 0, stream>>>(Wq, WqT, DQ_, INNER_);
  wtrans_kernel<<<(DC_ * INNER_ + 255) / 256, 256, 0, stream>>>(Wk, WkT, DC_, INNER_);
  wtrans_kernel<<<(DC_ * INNER_ + 255) / 256, 256, 0, stream>>>(Wv, WvT, DC_, INNER_);
  wtrans_kernel<<<(INNER_ * DQ_ + 255) / 256, 256, 0, stream>>>(Wo, WoT, INNER_, DQ_);

  // 2) projections (16x64 tiles per wave)
  qproj_kernel<<<dim3(B_ * L_ / 16, INNER_ / 64), 32, 0, stream>>>(x, WqT, Qh);
  kvproj_kernel<<<dim3(B_ * S_ / 16, INNER_ / 64, 2), 32, 0, stream>>>(ctx, WkT, WvT, Kh, Vt);

  // 3) fused flash attention (8 query strips per block, async double-buffered K/V)
  attn_kernel<<<dim3(L_ / 16 / AWAVES_, B_ * H_), 32 * AWAVES_, 0, stream>>>(Qh, Kh, Vt, AO);

  // 4) output projection + bias + mask
  oproj_kernel<<<dim3(B_ * L_ / 16, DQ_ / 64), 32, 0, stream>>>(AO, WoT, bo, seq_lens, out);
}
